// MultiHeadAttention_42640435315371
// MI455X (gfx1250) — matrix operand order repaired and checked by reading, not compiled, not run on hardware
//
#include <hip/hip_runtime.h>

// ---------- problem constants (from reference) ----------
#define NN      4
#define TT      4096
#define DMODEL  1024
#define HEADS   16
#define DHEAD   64
#define TOKENS  (NN * TT)          // 16384

// ---------- types ----------
typedef __bf16 bf16;
typedef __attribute__((ext_vector_type(8)))  __bf16 v8bf;   // 16 bytes
typedef __attribute__((ext_vector_type(16))) __bf16 v16bf;  // 32 bytes (WMMA A/B operand)
typedef __attribute__((ext_vector_type(8)))  float  v8f;    // WMMA C/D operand
typedef __attribute__((ext_vector_type(4)))  __bf16 v4bf;   // 8 bytes

static __device__ __forceinline__ v16bf join16(v8bf a, v8bf b) {
  return __builtin_shufflevector(a, b, 0,1,2,3,4,5,6,7,8,9,10,11,12,13,14,15);
}

static __device__ __forceinline__ v16bf cvt_f32x16(float4 f0, float4 f1,
                                                   float4 f2, float4 f3) {
  v16bf r;
  r[0]=(bf16)f0.x;  r[1]=(bf16)f0.y;  r[2]=(bf16)f0.z;  r[3]=(bf16)f0.w;
  r[4]=(bf16)f1.x;  r[5]=(bf16)f1.y;  r[6]=(bf16)f1.z;  r[7]=(bf16)f1.w;
  r[8]=(bf16)f2.x;  r[9]=(bf16)f2.y;  r[10]=(bf16)f2.z; r[11]=(bf16)f2.w;
  r[12]=(bf16)f3.x; r[13]=(bf16)f3.y; r[14]=(bf16)f3.z; r[15]=(bf16)f3.w;
  return r;
}

// low 32 bits of a flat shared-aperture address == LDS byte offset
static __device__ __forceinline__ unsigned lds_addr(const void* p) {
  return (unsigned)(uintptr_t)p;
}

// ---------- weight f32 -> bf16 convert ----------
__global__ __launch_bounds__(256)
void cvt_w_kernel(const float* __restrict__ W, bf16* __restrict__ O) {
  const size_t i = ((size_t)blockIdx.x * 256 + threadIdx.x) * 4;
  float4 f = *(const float4*)(W + i);
  v4bf r;
  r[0]=(bf16)f.x; r[1]=(bf16)f.y; r[2]=(bf16)f.z; r[3]=(bf16)f.w;
  *(v4bf*)(O + i) = r;
}

// ---------- GEMM: C(M=TOKENS x N=DMODEL) = A(M x K) * W^T, K=DMODEL ----------
// A_F32 : A is f32 (convert per-fragment) else bf16.
// OUTK  : 0 -> bf16 out, per-token (d,h) permuted columns (for Qt / Kt)
//         1 -> bf16 out, natural columns (for Vp)
//         2 -> f32 out, natural (final output)
// Block: 256 threads = 8 waves; block tile 64(M) x 256(N); wave tile 32 x 64.
template<int A_F32, int OUTK>
__global__ __launch_bounds__(256)
void gemm_bf16_kernel(const float* __restrict__ Af, const bf16* __restrict__ Ab,
                      const bf16* __restrict__ Bw,  bf16* __restrict__ Cb,
                      float* __restrict__ Cf) {
  const int lane = threadIdx.x & 31;
  const int w    = threadIdx.x >> 5;
  const int n16  = lane & 15;
  const int hf   = lane >> 4;                 // 0 or 1
  const int rm   = blockIdx.x * 64  + (w & 1) * 32;   // wave M base
  const int cn   = blockIdx.y * 256 + (w >> 1) * 64;  // wave N base
  const int abase = hf * 8;                   // A k-chunk base within 16
  const int bbase = hf * 8;                   // B k-chunk base within 16

  v8f acc[2][4] = {};

  for (int kk = 0; kk < DMODEL; kk += 32) {
    v16bf a[2];
    #pragma unroll
    for (int i = 0; i < 2; ++i) {
      const size_t row = (size_t)(rm + i * 16 + n16);
      if constexpr (A_F32) {
        const float* ap = Af + row * DMODEL + kk;
        float4 f0 = *(const float4*)(ap + abase);
        float4 f1 = *(const float4*)(ap + abase + 4);
        float4 f2 = *(const float4*)(ap + 16 + abase);
        float4 f3 = *(const float4*)(ap + 16 + abase + 4);
        a[i] = cvt_f32x16(f0, f1, f2, f3);
        if (kk + 32 < DMODEL) __builtin_prefetch(ap + 32, 0, 1);  // next K-step A
      } else {
        const bf16* ap = Ab + row * DMODEL + kk;
        a[i] = join16(*(const v8bf*)(ap + abase),
                      *(const v8bf*)(ap + 16 + abase));
        if (kk + 32 < DMODEL) __builtin_prefetch(ap + 32, 0, 1);
      }
    }
    #pragma unroll
    for (int j = 0; j < 4; ++j) {
      const size_t col = (size_t)(cn + j * 16 + n16);
      const bf16* bp = Bw + col * DMODEL + kk + bbase;   // W row 'col', contiguous K
      v16bf b = join16(*(const v8bf*)bp, *(const v8bf*)(bp + 16));
      #pragma unroll
      for (int i = 0; i < 2; ++i)
        acc[i][j] = __builtin_amdgcn_wmma_f32_16x16x32_bf16(
            false, a[i], false, b, (short)0, acc[i][j], false, false);
    }
  }

  #pragma unroll
  for (int i = 0; i < 2; ++i)
    #pragma unroll
    for (int j = 0; j < 4; ++j)
      #pragma unroll
      for (int g = 0; g < 8; ++g) {
        const size_t row = (size_t)(rm + i * 16 + g + hf * 8);
        const int    col = cn + j * 16 + n16;
        const float  v   = acc[i][j][g];
        if constexpr (OUTK == 0) {
          const int colp = ((col & 63) << 4) | (col >> 6);  // (h*64+d) -> d*16+h
          Cb[row * DMODEL + colp] = (bf16)v;
        } else if constexpr (OUTK == 1) {
          Cb[row * DMODEL + col] = (bf16)v;
        } else {
          Cf[row * DMODEL + col] = v;
        }
      }
}

// ---------- per-token attention ----------
// One block = one token, 4 waves; wave s handles d1 strip [s*16, s*16+16).
// energy(64x64) = Qh(64x16) @ Kh(16x64) / 32 ; softmax rows ; head = attn @ Vh(64x16)
// Qt/Kt rows are per-token (d,h) row-major; Vp rows natural (h,d).
// K-row and V-row (2 KB each) are staged into LDS with CDNA5 async loads.
__global__ __launch_bounds__(128)
void attn_kernel(const bf16* __restrict__ Qt, const bf16* __restrict__ Kt,
                 const bf16* __restrict__ Vp, const int* __restrict__ mask,
                 bf16* __restrict__ Cc) {
  __shared__ float lds_e[4][16 * 64];
  __shared__ bf16  lds_a[4][16 * 64];
  __shared__ bf16  kt_lds[DMODEL];   // token's Kt row, (d,h) layout
  __shared__ bf16  vp_lds[DMODEL];   // token's Vp row, (h,d) layout

  const int tk    = blockIdx.x;
  const int tid   = threadIdx.x;
  const int s     = tid >> 5;
  const int lane  = tid & 31;
  const int n16   = lane & 15;
  const int hf    = lane >> 4;
  const int abase = hf * 8;
  const size_t rowoff = (size_t)tk * DMODEL;

  // ---- async stage: 128 threads x 16 B == one 2 KB row per issue ----
  {
    const unsigned kdst = lds_addr(&kt_lds[0]) + tid * 16;
    const unsigned vdst = lds_addr(&vp_lds[0]) + tid * 16;
    const bf16* kg = Kt + rowoff + tid * 8;
    const bf16* vg = Vp + rowoff + tid * 8;
    asm volatile("global_load_async_to_lds_b128 %0, %1, off"
                 :: "v"(kdst), "v"(kg) : "memory");
    asm volatile("global_load_async_to_lds_b128 %0, %1, off"
                 :: "v"(vdst), "v"(vg) : "memory");
    asm volatile("s_wait_asynccnt 0x0" ::: "memory");
  }
  __syncthreads();

  // A fragment: M=d1 strip rows, K=heads (16 valid, upper 16 zero-padded)
  v8bf z8 = {};
  v16bf a = join16(*(const v8bf*)(Qt + rowoff + (s * 16 + n16) * 16 + abase), z8);

  v8f e[4] = {};
  #pragma unroll
  for (int j = 0; j < 4; ++j) {
    v16bf b;
    {  // elements 8-15 carry K=16..31 which is zero-padded
      const bf16* bp = &kt_lds[(j * 16 + n16) * 16 + abase];
      b = join16(*(const v8bf*)bp, z8);
    }
    e[j] = __builtin_amdgcn_wmma_f32_16x16x32_bf16(
        false, a, false, b, (short)0, e[j], false, false);
  }

  // energy strip -> LDS (scaled by 1/32 per reference's d_head/2 bug)
  #pragma unroll
  for (int j = 0; j < 4; ++j)
    #pragma unroll
    for (int g = 0; g < 8; ++g)
      lds_e[s][(g + hf * 8) * 64 + j * 16 + n16] = e[j][g] * 0.03125f;

  // softmax: lane owns row r = n16, half hf (32 cols). DS ops within a wave
  // are in-order, so the write->read through LDS needs no barrier.
  const int   mval = mask[tk];
  float*      rp = &lds_e[s][n16 * 64 + hf * 32];
  bf16*       aw = &lds_a[s][n16 * 64 + hf * 32];
  if (mval == 0) {
    const bf16 u = (bf16)(1.0f / 64.0f);   // softmax of all -1e20 is uniform
    #pragma unroll
    for (int i = 0; i < 32; ++i) aw[i] = u;
  } else {
    float vals[32];
    #pragma unroll
    for (int i = 0; i < 32; ++i) vals[i] = rp[i];
    float mx = -3.0e38f;
    #pragma unroll
    for (int i = 0; i < 32; ++i) mx = fmaxf(mx, vals[i]);
    mx = fmaxf(mx, __shfl_xor(mx, 16, 32));
    float sum = 0.0f;
    #pragma unroll
    for (int i = 0; i < 32; ++i) { vals[i] = __expf(vals[i] - mx); sum += vals[i]; }
    sum += __shfl_xor(sum, 16, 32);
    const float inv = 1.0f / sum;
    #pragma unroll
    for (int i = 0; i < 32; ++i) aw[i] = (bf16)(vals[i] * inv);
  }

  // head strip: attn(16x64) @ Vh(64x16), two K=32 steps
  v8f h = {};
  #pragma unroll
  for (int st = 0; st < 2; ++st) {
    const bf16* lp = &lds_a[s][n16 * 64 + st * 32 + abase];
    v16bf a2 = join16(*(const v8bf*)lp, *(const v8bf*)(lp + 16));
    const bf16* vp = &vp_lds[n16 * 64 + st * 32 + abase];  // col h contiguous in k
    v16bf b2 = join16(*(const v8bf*)vp, *(const v8bf*)(vp + 16));
    h = __builtin_amdgcn_wmma_f32_16x16x32_bf16(
        false, a2, false, b2, (short)0, h, false, false);
  }

  // concat[j] with j = d*16 + h
  #pragma unroll
  for (int g = 0; g < 8; ++g) {
    const int d = s * 16 + g + hf * 8;
    Cc[rowoff + d * 16 + n16] = (bf16)h[g];
  }
}

// ---------- host-side orchestration ----------
extern "C" void kernel_launch(void* const* d_in, const int* in_sizes, int n_in,
                              void* d_out, int out_size, void* d_ws, size_t ws_size,
                              hipStream_t stream) {
  const float* q  = (const float*)d_in[0];
  const float* k  = (const float*)d_in[1];
  const float* v  = (const float*)d_in[2];
  const int*   mk = (const int*)d_in[3];
  const float* Wq = (const float*)d_in[4];
  const float* Wk = (const float*)d_in[5];
  const float* Wv = (const float*)d_in[6];
  const float* Wo = (const float*)d_in[7];
  float* out = (float*)d_out;

  char* ws = (char*)d_ws;
  const size_t wsz  = (size_t)DMODEL * DMODEL * sizeof(bf16);  // 2 MB each
  const size_t asz  = (size_t)TOKENS * DMODEL * sizeof(bf16);  // 32 MB each
  bf16* Wqb = (bf16*)(ws + 0 * wsz);
  bf16* Wkb = (bf16*)(ws + 1 * wsz);
  bf16* Wvb = (bf16*)(ws + 2 * wsz);
  bf16* Wob = (bf16*)(ws + 3 * wsz);
  bf16* Qt  = (bf16*)(ws + 4 * wsz + 0 * asz);
  bf16* Kt  = (bf16*)(ws + 4 * wsz + 1 * asz);
  bf16* Vp  = (bf16*)(ws + 4 * wsz + 2 * asz);
  bf16* Cc  = (bf16*)(ws + 4 * wsz + 3 * asz);

  const dim3 cg((DMODEL * DMODEL) / (256 * 4));
  cvt_w_kernel<<<cg, 256, 0, stream>>>(Wq, Wqb);
  cvt_w_kernel<<<cg, 256, 0, stream>>>(Wk, Wkb);
  cvt_w_kernel<<<cg, 256, 0, stream>>>(Wv, Wvb);
  cvt_w_kernel<<<cg, 256, 0, stream>>>(Wo, Wob);

  const dim3 gg(TOKENS / 64, DMODEL / 256);
  gemm_bf16_kernel<1, 0><<<gg, 256, 0, stream>>>(q, nullptr, Wqb, Qt, nullptr);
  gemm_bf16_kernel<1, 0><<<gg, 256, 0, stream>>>(k, nullptr, Wkb, Kt, nullptr);
  gemm_bf16_kernel<1, 1><<<gg, 256, 0, stream>>>(v, nullptr, Wvb, Vp, nullptr);

  attn_kernel<<<TOKENS, 128, 0, stream>>>(Qt, Kt, Vp, mk, Cc);

  gemm_bf16_kernel<0, 2><<<gg, 256, 0, stream>>>(nullptr, Cc, Wob, nullptr, out);

  (void)in_sizes; (void)n_in; (void)out_size; (void)ws_size;
}
